// SlotAttention_21337397527139
// MI455X (gfx1250) — matrix operand order repaired and checked by reading, not compiled, not run on hardware
//
#include <hip/hip_runtime.h>
#include <hip/hip_bf16.h>

// ---------------- problem constants ----------------
constexpr int Bsz = 64;
constexpr int Nn  = 4096;
constexpr int Dd  = 256;
constexpr int Ks  = 8;      // slots
constexpr int IT  = 3;

typedef __attribute__((ext_vector_type(16))) __bf16       v16bf;
typedef __attribute__((ext_vector_type(8)))  __bf16       v8bf;
typedef __attribute__((ext_vector_type(8)))  float        v8f;
typedef __attribute__((ext_vector_type(4)))  float        v4f;
typedef __attribute__((ext_vector_type(4)))  unsigned int v4u;
typedef __attribute__((ext_vector_type(8)))  int          v8i;
typedef __attribute__((ext_vector_type(4)))  int          v4i;

#if defined(__has_builtin)
#if __has_builtin(__builtin_amdgcn_tensor_load_to_lds) && \
    __has_builtin(__builtin_amdgcn_s_wait_tensorcnt)
#define HAVE_TDM 1
#endif
#if __has_builtin(__builtin_amdgcn_sched_barrier)
#define SCHED_FENCE() __builtin_amdgcn_sched_barrier(0)
#endif
#endif
#ifndef HAVE_TDM
#define HAVE_TDM 0
#endif
#ifndef SCHED_FENCE
#define SCHED_FENCE()
#endif

// -------- WMMA fragment loaders (gfx1250 wave32 layouts, CDNA5 ISA) --------
// A 16x32 bf16: lane L -> row M=L&15; V0-3: K=(L>>4)*8+0..7, V4-7: K=16+(L>>4)*8+0..7
__device__ inline v16bf load_a_frag_lds(const __bf16* base, int rowStride, int rm0, int kb) {
  int l = threadIdx.x & 31, m = l & 15, hid = l >> 4;
  const __bf16* p0 = base + (size_t)(rm0 + m) * rowStride + kb + hid * 8;
  v8bf lo = *(const v8bf*)p0;
  v8bf hi = *(const v8bf*)(p0 + 16);
  v16bf a;
#pragma unroll
  for (int i = 0; i < 8; ++i) { a[i] = lo[i]; a[8 + i] = hi[i]; }
  return a;
}
__device__ inline v16bf load_a_frag_g(const __bf16* A, int ld, int rm0, int kb) {
  int l = threadIdx.x & 31, m = l & 15, hid = l >> 4;
  const __bf16* p0 = A + (size_t)(rm0 + m) * ld + kb + hid * 8;
  v8bf lo = *(const v8bf*)p0;
  v8bf hi = *(const v8bf*)(p0 + 16);
  v16bf a;
#pragma unroll
  for (int i = 0; i < 8; ++i) { a[i] = lo[i]; a[8 + i] = hi[i]; }
  return a;
}
// B 32x16 bf16: lane L -> col N=L&15; V0-3: K=(L>>4)*8+0..7, V4-7: K=16+(L>>4)*8+0..7
__device__ inline v16bf load_b_frag(const __bf16* Bm, int ld, int ib0, int ob0) {
  int l = threadIdx.x & 31, n = l & 15, hid = l >> 4;
  const __bf16* p = Bm + (size_t)(ib0 + hid * 8) * ld + ob0 + n;
  v16bf bv;
#pragma unroll
  for (int i = 0; i < 8; ++i) { bv[i] = p[(size_t)i * ld]; bv[8 + i] = p[(size_t)(16 + i) * ld]; }
  return bv;
}

__device__ inline float wave_sum(float v) {
#pragma unroll
  for (int m = 16; m >= 1; m >>= 1) v += __shfl_xor(v, m, 32);
  return v;
}

#if HAVE_TDM
// Build a 2D D# (CDNA5 ISA) and issue TENSOR_LOAD_TO_LDS.
// Tensor == tile: cols x rows of f32, row stride rowStride elements.
__device__ inline void tdm_load_tile_f32(unsigned lds_off, const float* gsrc,
                                         int rows, int cols, int rowStride) {
  unsigned long long ga = (unsigned long long)(size_t)gsrc;
  v4u g0;
  g0[0] = 1u;                                   // count=1 user descriptor
  g0[1] = lds_off;                              // lds_addr (bytes)
  g0[2] = (unsigned)ga;                         // global_addr[31:0]
  g0[3] = (unsigned)((ga >> 32) & 0x1FFFFFFu)   // global_addr[56:32]
        | (2u << 30);                           // type = 2 ("image")
  v8i g1;
  g1[0] = (int)(2u << 16);                      // data_size = 2 (4 bytes)
  g1[1] = (int)(((unsigned)cols & 0xFFFFu) << 16);            // tensor_dim0[15:0] @bit48
  g1[2] = (int)(((unsigned)cols >> 16) |
                (((unsigned)rows & 0xFFFFu) << 16));          // dim0 hi | tensor_dim1 lo
  g1[3] = (int)(((unsigned)rows >> 16) |
                (((unsigned)cols & 0xFFFFu) << 16));          // dim1 hi | tile_dim0
  g1[4] = (int)((unsigned)rows & 0xFFFFu);                    // tile_dim1 (tile_dim2=0)
  g1[5] = (int)(unsigned)rowStride;                           // tensor_dim0_stride lo32
  g1[6] = 0;
  g1[7] = 0;
  v4i z4 = {0, 0, 0, 0};
#if __has_include(<hip/amd_detail/amd_gfx1250_TDM.h>)
  v8i z8 = {0, 0, 0, 0, 0, 0, 0, 0};
  __builtin_amdgcn_tensor_load_to_lds(g0, g1, z4, z4, z8, 0);   // 6-arg toolchain
#else
  __builtin_amdgcn_tensor_load_to_lds(g0, g1, z4, z4, 0);       // 5-arg ROCm 7.2
#endif
}
#endif

// ---------------- prep: transposed weight copies ----------------
__global__ __launch_bounds__(256) void k_prep_w(
    const float* __restrict__ Wk, const float* __restrict__ Wv,
    const float* __restrict__ Wih, const float* __restrict__ Whh,
    const float* __restrict__ W1, const float* __restrict__ W2,
    __bf16* __restrict__ WkT, __bf16* __restrict__ WvT,
    float* __restrict__ WihT, float* __restrict__ WhhT,
    float* __restrict__ W1T, float* __restrict__ W2T) {
  int idx = blockIdx.x * 256 + threadIdx.x;
  if (idx < Dd * Dd) {
    int i = idx / Dd, o = idx % Dd;
    WkT[idx] = (__bf16)Wk[(size_t)o * Dd + i];
    WvT[idx] = (__bf16)Wv[(size_t)o * Dd + i];
  }
  if (idx < Dd * 3 * Dd) {
    int i = idx / (3 * Dd), j = idx % (3 * Dd);
    WihT[idx] = Wih[(size_t)j * Dd + i];
    WhhT[idx] = Whh[(size_t)j * Dd + i];
  }
  if (idx < Dd * 2 * Dd) {
    int i = idx / (2 * Dd), j = idx % (2 * Dd);
    W1T[idx] = W1[(size_t)j * Dd + i];
    int jj = idx / Dd, d2 = idx % Dd;
    W2T[idx] = W2[(size_t)d2 * 2 * Dd + jj];
  }
}

// ---------------- slot init: LN(slot_inits) broadcast over B ----------------
__global__ __launch_bounds__(256) void k_slot_init(
    const float* __restrict__ sinit, const float* __restrict__ g,
    const float* __restrict__ bb, float* __restrict__ slots) {
  int kk = blockIdx.x, t = threadIdx.x;
  __shared__ float red[16];
  float x = sinit[kk * Dd + t];
  float s = wave_sum(x), s2 = wave_sum(x * x);
  int w = t >> 5;
  if ((t & 31) == 0) { red[w] = s; red[8 + w] = s2; }
  __syncthreads();
  s = 0.f; s2 = 0.f;
#pragma unroll
  for (int i = 0; i < 8; ++i) { s += red[i]; s2 += red[8 + i]; }
  float mu = s / Dd, var = s2 / Dd - mu * mu;
  float val = (x - mu) * rsqrtf(var + 1e-5f) * g[t] + bb[t];
  for (int b = 0; b < Bsz; ++b)
    slots[((size_t)b * Ks + kk) * Dd + t] = val;
}

// ---------- fused LN(x) -> bf16 -> WMMA k,v, TDM double-buffered input ----------
// kT [B][D][N] bf16 (B-operand for logits), v [B][N][D] bf16 (B-operand for updates).
__global__ __launch_bounds__(256) void k_ln_kv(
    const float* __restrict__ inp, const float* __restrict__ lng,
    const float* __restrict__ lnb,
    const __bf16* __restrict__ WkT, const __bf16* __restrict__ WvT,
    __bf16* __restrict__ kT, __bf16* __restrict__ vmat) {
  constexpr int TN = 64;    // rows (n) per chunk
  constexpr int CH = 4;     // chunks per block (TDM pipeline depth = 2 buffers)
  constexpr int RS = 264;   // padded bf16 LDS row stride
  __shared__ float  xsf[2][TN * Dd];   // f32 staging (TDM destination)
  __shared__ __bf16 xs[TN * RS];       // normalized bf16 tile (WMMA A source)

  int b      = blockIdx.x;
  int n0base = blockIdx.y * (TN * CH);
  int wave   = threadIdx.x >> 5;
  int l      = threadIdx.x & 31;
  int rt     = wave & 3;               // row-tile 0..3 (16 rows each)
  int mat    = wave >> 2;              // 0 -> k, 1 -> v
  int rm0    = rt * 16;
  const __bf16* W = mat ? WvT : WkT;

#if HAVE_TDM
  if (wave == 0)
    tdm_load_tile_f32((unsigned)(size_t)&xsf[0][0],
                      inp + ((size_t)b * Nn + n0base) * Dd, TN, Dd, Dd);
#endif

  for (int ch = 0; ch < CH; ++ch) {
    int buf = ch & 1;
    int n0  = n0base + ch * TN;
#if HAVE_TDM
    if (wave == 0) {     // prefetch next chunk, then wait for current
      if (ch + 1 < CH) {
        tdm_load_tile_f32((unsigned)(size_t)&xsf[buf ^ 1][0],
                          inp + ((size_t)b * Nn + n0 + TN) * Dd, TN, Dd, Dd);
        __builtin_amdgcn_s_wait_tensorcnt(1);
      } else {
        __builtin_amdgcn_s_wait_tensorcnt(0);
      }
    }
#else
    // fallback: cooperative copy global -> LDS staging
    {
      const float* g = inp + ((size_t)b * Nn + n0) * Dd;
#pragma unroll
      for (int i = 0; i < 16; ++i) {
        int off = i * 1024 + threadIdx.x * 4;
        *(v4f*)&xsf[buf][off] = *(const v4f*)(g + off);
      }
    }
#endif
    __syncthreads();   // releases TDM-filled buffer to all waves

    // Phase 1: LayerNorm 8 rows per wave from LDS staging, write bf16 tile.
#pragma unroll
    for (int rr = 0; rr < 8; ++rr) {
      int r = wave * 8 + rr;
      const float* row = &xsf[buf][r * Dd];
      v4f x0 = *(const v4f*)(row + l * 8);
      v4f x1 = *(const v4f*)(row + l * 8 + 4);
      float s = 0.f, s2 = 0.f;
#pragma unroll
      for (int i = 0; i < 4; ++i) {
        s += x0[i] + x1[i];
        s2 += x0[i] * x0[i] + x1[i] * x1[i];
      }
      s = wave_sum(s); s2 = wave_sum(s2);
      float mu  = s * (1.0f / Dd);
      float inv = rsqrtf(s2 * (1.0f / Dd) - mu * mu + 1e-5f);
      v8bf pk;
#pragma unroll
      for (int i = 0; i < 4; ++i) {
        int c0 = l * 8 + i, c1 = l * 8 + 4 + i;
        pk[i]     = (__bf16)((x0[i] - mu) * inv * lng[c0] + lnb[c0]);
        pk[4 + i] = (__bf16)((x1[i] - mu) * inv * lng[c1] + lnb[c1]);
      }
      *(v8bf*)(xs + r * RS + l * 8) = pk;
    }
    __syncthreads();

    // Phase 2: WMMA. Preload A once per chunk; per tile, all 8 B loads issue as
    // one clause, SCHED_FENCE keeps the scheduler from re-sinking them into the
    // wmma chain (waits relax from loadcnt==0 to a descending in-order pattern).
    v16bf afr[8];
#pragma unroll
    for (int kc = 0; kc < 8; ++kc) afr[kc] = load_a_frag_lds(xs, RS, rm0, kc * 32);
    SCHED_FENCE();

    for (int ot = 0; ot < 16; ++ot) {
      v16bf bfr[8];
#pragma unroll
      for (int kc = 0; kc < 8; ++kc) bfr[kc] = load_b_frag(W, Dd, kc * 32, ot * 16);
      SCHED_FENCE();          // loads stay above, wmma chain below
      v8f c = {};
#pragma unroll
      for (int kc = 0; kc < 8; ++kc)
        c = __builtin_amdgcn_wmma_f32_16x16x32_bf16(false, afr[kc], false, bfr[kc],
                                                    (short)0, c, false, false);
      int o  = ot * 16 + (l & 15);
      int nb = n0 + rm0 + (l >> 4) * 8;
      if (mat == 0) {   // kT[b][o][nb..nb+8) -> one b128 store
        v8bf pk;
#pragma unroll
        for (int r = 0; r < 8; ++r) pk[r] = (__bf16)c[r];
        *(v8bf*)(kT + (((size_t)b * Dd + o) * Nn + nb)) = pk;
      } else {          // v[b][n][o]
#pragma unroll
        for (int r = 0; r < 8; ++r)
          vmat[((size_t)b * Nn + nb + r) * Dd + o] = (__bf16)c[r];
      }
    }
    __syncthreads();   // xs / staging reuse fence before next chunk
  }
}

// ---------------- q projection (tiny, f32 VALU), padded to 16 rows bf16 ----------------
__global__ __launch_bounds__(256) void k_qproj(
    const float* __restrict__ slots, const float* __restrict__ Wq,
    __bf16* __restrict__ qb) {
  int idx = blockIdx.x * 256 + threadIdx.x;   // B * 16 * D
  int o  = idx % Dd;
  int kk = (idx / Dd) & 15;
  int b  = idx / (Dd * 16);
  float acc = 0.f;
  if (kk < Ks) {
    const float* sr = slots + ((size_t)b * Ks + kk) * Dd;
    const float* wr = Wq + (size_t)o * Dd;
    for (int i = 0; i < Dd; ++i) acc += sr[i] * wr[i];
  }
  qb[idx] = (__bf16)acc;   // rows 8..15 -> 0 (WMMA pad)
}

// ---------------- logits WMMA + in-register softmax over slots ----------------
__global__ __launch_bounds__(256) void k_logits(
    const __bf16* __restrict__ qb, const __bf16* __restrict__ kT,
    __bf16* __restrict__ attnb, float* __restrict__ out_attn, int write_out) {
  int b    = blockIdx.x;
  int wave = threadIdx.x >> 5;
  int l    = threadIdx.x & 31;
  int nb0  = (blockIdx.y * 8 + wave) * 16;
  const __bf16* q   = qb + (size_t)b * 16 * Dd;
  const __bf16* kTb = kT + (size_t)b * Dd * Nn;
  // preload both operand sets; fence so the 16 loads issue ahead of the chain
  v16bf afr[8], bfr[8];
#pragma unroll
  for (int kc = 0; kc < 8; ++kc) afr[kc] = load_a_frag_g(q, Dd, 0, kc * 32);
#pragma unroll
  for (int kc = 0; kc < 8; ++kc) bfr[kc] = load_b_frag(kTb, Nn, kc * 32, nb0);
  SCHED_FENCE();
  v8f c = {};
#pragma unroll
  for (int kc = 0; kc < 8; ++kc)
    c = __builtin_amdgcn_wmma_f32_16x16x32_bf16(false, afr[kc], false, bfr[kc],
                                                (short)0, c, false, false);
  int n = nb0 + (l & 15);
  if (l < 16) {
    // lane l<16 holds all 8 slot logits of column n -> softmax fully in-register
    const float scale = 0.0625f;   // D^-0.5
    float e[Ks], mx = -3.0e38f;
#pragma unroll
    for (int r = 0; r < Ks; ++r) { e[r] = c[r] * scale; mx = fmaxf(mx, e[r]); }
    float s = 0.f;
#pragma unroll
    for (int r = 0; r < Ks; ++r) { e[r] = __expf(e[r] - mx); s += e[r]; }
    float inv = 1.0f / s;
    v8f a8;
#pragma unroll
    for (int r = 0; r < Ks; ++r) {
      float a = e[r] * inv;
      a8[r] = a;
      attnb[((size_t)b * 16 + r) * Nn + n] = (__bf16)a;
    }
    if (write_out)   // attn_weights [B][N][K]: one 32B store
      *(v8f*)(out_attn + ((size_t)b * Nn + n) * Ks) = a8;
  } else {
#pragma unroll
    for (int r = 0; r < Ks; ++r)
      attnb[((size_t)b * 16 + Ks + r) * Nn + n] = (__bf16)0.f;
  }
}

// ------- updates = attn @ v  (WMMA over inner N=4096, software-pipelined) -------
__global__ __launch_bounds__(256) void k_updates(
    const __bf16* __restrict__ attnb, const __bf16* __restrict__ vmat,
    float* __restrict__ updates) {
  int b    = blockIdx.x;
  int wave = threadIdx.x >> 5;
  int l    = threadIdx.x & 31;
  const __bf16* A = attnb + (size_t)b * 16 * Nn;
  const __bf16* V = vmat + (size_t)b * Nn * Dd;
  int ot0 = wave * 2, ot1 = ot0 + 1;
  v8f c0 = {}, c1 = {};
  // double-buffered pipeline: fragments for kc+1 stay in flight during wmma of kc
  v16bf a  = load_a_frag_g(A, Nn, 0, 0);
  v16bf b0 = load_b_frag(V, Dd, 0, ot0 * 16);
  v16bf b1 = load_b_frag(V, Dd, 0, ot1 * 16);
  for (int kc = 0; kc < Nn / 32 - 1; ++kc) {
    if ((kc & 7) == 0)   // stream-ahead hint on the 2 MB v matrix
      __builtin_prefetch(V + (size_t)((kc + 8) * 32 + l) * Dd, 0, 1);
    v16bf an  = load_a_frag_g(A, Nn, 0, (kc + 1) * 32);
    v16bf b0n = load_b_frag(V, Dd, (kc + 1) * 32, ot0 * 16);
    v16bf b1n = load_b_frag(V, Dd, (kc + 1) * 32, ot1 * 16);
    SCHED_FENCE();       // keep next-step loads above current-step wmmas
    c0 = __builtin_amdgcn_wmma_f32_16x16x32_bf16(false, a, false, b0,
                                                 (short)0, c0, false, false);
    c1 = __builtin_amdgcn_wmma_f32_16x16x32_bf16(false, a, false, b1,
                                                 (short)0, c1, false, false);
    a = an; b0 = b0n; b1 = b1n;
  }
  c0 = __builtin_amdgcn_wmma_f32_16x16x32_bf16(false, a, false, b0,
                                               (short)0, c0, false, false);
  c1 = __builtin_amdgcn_wmma_f32_16x16x32_bf16(false, a, false, b1,
                                               (short)0, c1, false, false);
  if ((l >> 4) == 0) {   // lanes 0..15 hold rows 0..7 (real slots)
    int o0 = ot0 * 16 + (l & 15), o1 = ot1 * 16 + (l & 15);
#pragma unroll
    for (int r = 0; r < 8; ++r) {
      updates[((size_t)b * Ks + r) * Dd + o0] = c0[r];
      updates[((size_t)b * Ks + r) * Dd + o1] = c1[r];
    }
  }
}

// ---------------- fused GRU cell + LayerNorm + MLP residual (f32) ----------------
__global__ __launch_bounds__(256) void k_gru_mlp(
    const float* __restrict__ updates, float* __restrict__ slots,
    const float* __restrict__ WihT, const float* __restrict__ WhhT,
    const float* __restrict__ b_ih, const float* __restrict__ b_hh,
    const float* __restrict__ lng, const float* __restrict__ lnb,
    const float* __restrict__ W1T, const float* __restrict__ b1,
    const float* __restrict__ W2T, const float* __restrict__ b2,
    float* __restrict__ out_slots, int write_out) {
  __shared__ float xr[Dd], hr[Dd], hln[Dd], hid[2 * Dd], red[16];
  int bk = blockIdx.x;
  int b = bk / Ks, kk = bk % Ks;
  int t = threadIdx.x;
  size_t base = ((size_t)b * Ks + kk) * Dd;
  xr[t] = updates[base + t];
  hr[t] = slots[base + t];
  __syncthreads();

  float gi_r = b_ih[t], gi_z = b_ih[Dd + t], gi_n = b_ih[2 * Dd + t];
  float gh_r = b_hh[t], gh_z = b_hh[Dd + t], gh_n = b_hh[2 * Dd + t];
  for (int i = 0; i < Dd; ++i) {
    float xi = xr[i], hi = hr[i];
    const float* wi = WihT + (size_t)i * 3 * Dd;
    const float* wh = WhhT + (size_t)i * 3 * Dd;
    gi_r += xi * wi[t]; gi_z += xi * wi[Dd + t]; gi_n += xi * wi[2 * Dd + t];
    gh_r += hi * wh[t]; gh_z += hi * wh[Dd + t]; gh_n += hi * wh[2 * Dd + t];
  }
  float r  = 1.f / (1.f + __expf(-(gi_r + gh_r)));
  float z  = 1.f / (1.f + __expf(-(gi_z + gh_z)));
  float nn = tanhf(gi_n + r * gh_n);
  float hv = (1.f - z) * nn + z * hr[t];

  float s = wave_sum(hv), s2 = wave_sum(hv * hv);
  int w = t >> 5;
  if ((t & 31) == 0) { red[w] = s; red[8 + w] = s2; }
  __syncthreads();
  s = 0.f; s2 = 0.f;
#pragma unroll
  for (int i = 0; i < 8; ++i) { s += red[i]; s2 += red[8 + i]; }
  float mu = s / Dd, var = s2 / Dd - mu * mu;
  hln[t] = (hv - mu) * rsqrtf(var + 1e-5f) * lng[t] + lnb[t];
  __syncthreads();

  float h0 = b1[t], h1 = b1[Dd + t];
  for (int i = 0; i < Dd; ++i) {
    float v = hln[i];
    const float* wr = W1T + (size_t)i * 2 * Dd;
    h0 += v * wr[t]; h1 += v * wr[Dd + t];
  }
  hid[t] = fmaxf(h0, 0.f); hid[Dd + t] = fmaxf(h1, 0.f);
  __syncthreads();

  float o = hv + b2[t];
  for (int j = 0; j < 2 * Dd; ++j) o += hid[j] * W2T[(size_t)j * Dd + t];
  slots[base + t] = o;
  if (write_out) out_slots[base + t] = o;
}

// ---------------- host orchestration ----------------
extern "C" void kernel_launch(void* const* d_in, const int* in_sizes, int n_in,
                              void* d_out, int out_size, void* d_ws, size_t ws_size,
                              hipStream_t stream) {
  (void)in_sizes; (void)n_in; (void)out_size; (void)ws_size;
  const float* inputs   = (const float*)d_in[0];
  const float* sinit    = (const float*)d_in[1];
  const float* Wq       = (const float*)d_in[2];
  const float* Wk       = (const float*)d_in[3];
  const float* Wv       = (const float*)d_in[4];
  const float* Wih      = (const float*)d_in[5];
  const float* Whh      = (const float*)d_in[6];
  const float* b_ih     = (const float*)d_in[7];
  const float* b_hh     = (const float*)d_in[8];
  const float* ln_in_g  = (const float*)d_in[9];
  const float* ln_in_b  = (const float*)d_in[10];
  const float* ln_sl_g  = (const float*)d_in[11];
  const float* ln_sl_b  = (const float*)d_in[12];
  const float* ln_ml_g  = (const float*)d_in[13];
  const float* ln_ml_b  = (const float*)d_in[14];
  const float* W1       = (const float*)d_in[15];
  const float* b1       = (const float*)d_in[16];
  const float* W2       = (const float*)d_in[17];
  const float* b2       = (const float*)d_in[18];

  float* out_slots = (float*)d_out;                         // [B,K,D]
  float* out_attn  = (float*)d_out + (size_t)Bsz * Ks * Dd; // [B,N,K]

  char* ws = (char*)d_ws;
  size_t off = 0;
  auto carve = [&](size_t bytes) { char* p = ws + off; off += (bytes + 255) & ~255ULL; return p; };
  __bf16* kT    = (__bf16*)carve((size_t)Bsz * Dd * Nn * 2);   // 128 MB
  __bf16* vmat  = (__bf16*)carve((size_t)Bsz * Nn * Dd * 2);   // 128 MB
  __bf16* attnb = (__bf16*)carve((size_t)Bsz * 16 * Nn * 2);   // 8 MB
  __bf16* qb    = (__bf16*)carve((size_t)Bsz * 16 * Dd * 2);
  float*  slots = (float*)carve((size_t)Bsz * Ks * Dd * 4);
  float*  upd   = (float*)carve((size_t)Bsz * Ks * Dd * 4);
  __bf16* WkT   = (__bf16*)carve((size_t)Dd * Dd * 2);
  __bf16* WvT   = (__bf16*)carve((size_t)Dd * Dd * 2);
  float*  WihT  = (float*)carve((size_t)Dd * 3 * Dd * 4);
  float*  WhhT  = (float*)carve((size_t)Dd * 3 * Dd * 4);
  float*  W1T   = (float*)carve((size_t)Dd * 2 * Dd * 4);
  float*  W2T   = (float*)carve((size_t)2 * Dd * Dd * 4);

  k_prep_w<<<(Dd * 3 * Dd + 255) / 256, 256, 0, stream>>>(
      Wk, Wv, Wih, Whh, W1, W2, WkT, WvT, WihT, WhhT, W1T, W2T);
  k_slot_init<<<Ks, 256, 0, stream>>>(sinit, ln_sl_g, ln_sl_b, slots);
  k_ln_kv<<<dim3(Bsz, Nn / 256), 256, 0, stream>>>(
      inputs, ln_in_g, ln_in_b, WkT, WvT, kT, vmat);

  for (int it = 0; it < IT; ++it) {
    int last = (it == IT - 1);
    k_qproj<<<(Bsz * 16 * Dd) / 256, 256, 0, stream>>>(slots, Wq, qb);
    k_logits<<<dim3(Bsz, Nn / 128), 256, 0, stream>>>(qb, kT, attnb, out_attn, last);
    k_updates<<<Bsz, 256, 0, stream>>>(attnb, vmat, upd);
    k_gru_mlp<<<Bsz * Ks, 256, 0, stream>>>(
        upd, slots, WihT, WhhT, b_ih, b_hh, ln_ml_g, ln_ml_b,
        W1T, b1, W2T, b2, out_slots, last);
  }
}
